// STEVESlotAttentionAdapter_80779744903468
// MI455X (gfx1250) — compile-verified, matrix operand order checked by reading, not run on hardware
//
#include <hip/hip_runtime.h>
#include <hip/hip_bf16.h>

// ---------------- problem constants ----------------
constexpr int kB = 32;        // batch
constexpr int kS = 7;         // slots
constexpr int kSP = 16;       // slots padded to WMMA tile
constexpr int kH = 128;       // slot dim
constexpr int kD = 768;       // feature dim
constexpr int kN = 4096;      // num features
constexpr int kMLP = 512;     // 4*H
constexpr float kEPS = 1e-8f;
constexpr float kQSCALE = 0.08838834764831845f; // H^-0.5

typedef __bf16 bf16_t;
typedef __attribute__((ext_vector_type(16))) __bf16 v16bf;
typedef __attribute__((ext_vector_type(8)))  __bf16 v8bf;
typedef __attribute__((ext_vector_type(4)))  __bf16 v4bf;
typedef __attribute__((ext_vector_type(8)))  float  v8f;

__device__ __forceinline__ bf16_t to_bf16(float f) { return (bf16_t)f; }

__device__ __forceinline__ v16bf concat8(v8bf lo, v8bf hi) {
  return __builtin_shufflevector(lo, hi, 0, 1, 2, 3, 4, 5, 6, 7,
                                 8, 9, 10, 11, 12, 13, 14, 15);
}

// workspace layout (bytes)
constexpr size_t OFF_K   = 0;                                  // k bf16 [B][N][H]
constexpr size_t SZ_K    = (size_t)kB * kN * kH * 2;
constexpr size_t OFF_VT  = OFF_K + SZ_K;                       // v^T bf16 [B][H][N]
constexpr size_t SZ_VT   = SZ_K;
constexpr size_t OFF_WK  = OFF_VT + SZ_VT;                     // Wk bf16 packed [D/32][H][32]
constexpr size_t SZ_W    = (size_t)kD * kH * 2;
constexpr size_t OFF_WV  = OFF_WK + SZ_W;                      // Wv bf16 packed
constexpr size_t OFF_Q   = OFF_WV + SZ_W;                      // q bf16 [B][16][H]
constexpr size_t SZ_Q    = (size_t)kB * kSP * kH * 2;
constexpr size_t OFF_AT  = OFF_Q + SZ_Q;                       // attn^T bf16 [B][16][N]
constexpr size_t SZ_AT   = (size_t)kB * kSP * kN * 2;
constexpr size_t OFF_CS  = OFF_AT + SZ_AT;                     // colsum f32 [B][16]
constexpr size_t SZ_CS   = (size_t)kB * kSP * 4;
constexpr size_t OFF_UPD = OFF_CS + SZ_CS;                     // updates f32 [B][16][H]
constexpr size_t SZ_UPD  = (size_t)kB * kSP * kH * 4;
constexpr size_t OFF_SL  = OFF_UPD + SZ_UPD;                   // slots_cur f32 [B][S][H]

// ---------------- K0: weight conversion + pack into WMMA-B fragment layout ------------
// packed layout: wpack[t][c][K], t = D-tile of 32, c = output column, K = 0..31.
// A lane (col c, chunk base kb in {0,8}) then loads chunks {kb..kb+7} and
// {16+kb..16+kb+7} as two contiguous 16B loads.
__global__ void k_wconv(const float* __restrict__ Wk, const float* __restrict__ Wv,
                        bf16_t* __restrict__ wkb, bf16_t* __restrict__ wvb) {
  int i = blockIdx.x * 256 + threadIdx.x;    // packed index
  if (i < kD * kH) {
    int K = i & 31;
    int c = (i >> 5) & (kH - 1);
    int t = i >> 12;                          // /(32*128)
    size_t src = (size_t)(t * 32 + K) * kH + c;
    wkb[i] = to_bf16(Wk[src] * kQSCALE);
    wvb[i] = to_bf16(Wv[src]);
  }
}

// ---------------- K1: fused LN(features) + k/v projection (bf16 WMMA) -----------------
// grid: B*N/16 blocks; 256 threads = 8 waves. Features read from HBM exactly once:
// coalesced float4 loads feed both the mean/var accumulation and an immediate bf16
// LDS stage; the normalize pass rewrites LDS in place.
__global__ __launch_bounds__(256) void k_ln_project(
    const float* __restrict__ features, const float* __restrict__ lng,
    const float* __restrict__ lnb, const bf16_t* __restrict__ wk,
    const bf16_t* __restrict__ wv, bf16_t* __restrict__ kout,
    bf16_t* __restrict__ vtout) {
  __shared__ __align__(16) __bf16 sf[16][kD];   // 24 KB staged rows (raw -> LN'd)
  __shared__ float reds[16][16], reds2[16][16];
  __shared__ float smean[16], srstd[16];

  const int tid = threadIdx.x;
  const int b  = (blockIdx.x * 16) / kN;
  const int n0 = (blockIdx.x * 16) % kN;
  const int row = tid >> 4, l16 = tid & 15;

  // single coalesced pass: accumulate stats + stage raw values as bf16
  {
    const float4* xr = (const float4*)(features + ((size_t)b * kN + n0 + row) * kD);
    float s = 0.f, s2 = 0.f;
#pragma unroll
    for (int ii = 0; ii < kD / 4 / 16; ii++) {     // 12 iterations
      int f4 = l16 + 16 * ii;
      float4 x = xr[f4];
      s  += x.x + x.y + x.z + x.w;
      s2 += x.x * x.x + x.y * x.y + x.z * x.z + x.w * x.w;
      v4bf xb;
      xb[0] = to_bf16(x.x); xb[1] = to_bf16(x.y);
      xb[2] = to_bf16(x.z); xb[3] = to_bf16(x.w);
      *(v4bf*)&sf[row][4 * f4] = xb;
    }
    reds[row][l16] = s; reds2[row][l16] = s2;
  }
  __syncthreads();
  if (l16 == 0) {
    float a = 0.f, c = 0.f;
    for (int j = 0; j < 16; j++) { a += reds[row][j]; c += reds2[row][j]; }
    float m = a / kD;
    float var = c / kD - m * m;
    smean[row] = m;
    srstd[row] = rsqrtf(var + 1e-5f);
  }
  __syncthreads();
  // normalize LDS in place (each element owned by exactly one thread)
  for (int idx = tid; idx < 16 * kD; idx += 256) {
    int r = idx / kD, j = idx - r * kD;
    float v = ((float)sf[r][j] - smean[r]) * srstd[r] * lng[j] + lnb[j];
    sf[r][j] = to_bf16(v);
  }
  __syncthreads();

  // WMMA GEMM over D
  const int wave = tid >> 5, lane = tid & 31;
  const int m16 = lane & 15;
  const int kb  = (lane < 16) ? 0 : 8;   // K-chunk base per A/B 16-bit layout
  const int ch  = kb >> 3;               // 0 or 1
  const int colw = wave * 16 + m16;
  v8f ck = {0.f, 0.f, 0.f, 0.f, 0.f, 0.f, 0.f, 0.f};
  v8f cv = ck;

  for (int kk = 0; kk < kD; kk += 32) {
    v16bf a = concat8(*(const v8bf*)&sf[m16][kk + kb],
                      *(const v8bf*)&sf[m16][kk + 16 + kb]);
    const v8bf* pk = (const v8bf*)(wk + (((size_t)(kk >> 5) * kH + colw) << 5));
    const v8bf* pv = (const v8bf*)(wv + (((size_t)(kk >> 5) * kH + colw) << 5));
    v16bf bk = concat8(pk[ch], pk[ch + 2]);
    v16bf bv = concat8(pv[ch], pv[ch + 2]);
    ck = __builtin_amdgcn_wmma_f32_16x16x32_bf16(false, a, false, bk, (short)0, ck, false, false);
    cv = __builtin_amdgcn_wmma_f32_16x16x32_bf16(false, a, false, bv, (short)0, cv, false, false);
  }

  const int rbase = 8 * (lane >> 4);
  // k row-major [B][N][H] (lanes 0..15 of a half-wave cover 32B of one row)
#pragma unroll
  for (int i = 0; i < 8; i++) {
    kout[((size_t)b * kN + n0 + rbase + i) * kH + colw] = to_bf16(ck[i]);
  }
  // v transposed [B][H][N]: per-lane 8 contiguous n -> one 16B store
  v8bf vv;
#pragma unroll
  for (int i = 0; i < 8; i++) vv[i] = to_bf16(cv[i]);
  *(v8bf*)&vtout[((size_t)b * kH + colw) * kN + n0 + rbase] = vv;
}

// ---------------- K2: q = LN(slots)@Wq (tiny, fp32 VALU) + zero colsum ----------------
__global__ __launch_bounds__(128) void k_qproj(
    const float* __restrict__ slots_cur, const float* __restrict__ lnsg,
    const float* __restrict__ lnsb, const float* __restrict__ Wq,
    bf16_t* __restrict__ qout, float* __restrict__ colsum) {
  const int b = blockIdx.x, t = threadIdx.x;
  __shared__ float sh[kH];
  __shared__ float red1[128], red2[128];
  if (t < kSP) colsum[b * kSP + t] = 0.f;

  for (int s = 0; s < kS; s++) {
    float x = slots_cur[((size_t)b * kS + s) * kH + t];
    red1[t] = x; red2[t] = x * x;
    __syncthreads();
    for (int off = 64; off > 0; off >>= 1) {
      if (t < off) { red1[t] += red1[t + off]; red2[t] += red2[t + off]; }
      __syncthreads();
    }
    float m = red1[0] / kH;
    float var = red2[0] / kH - m * m;
    float rstd = rsqrtf(var + 1e-5f);
    sh[t] = (x - m) * rstd * lnsg[t] + lnsb[t];
    __syncthreads();
    float acc = 0.f;
    for (int j = 0; j < kH; j++) acc += sh[j] * Wq[(size_t)j * kH + t];
    qout[((size_t)b * kSP + s) * kH + t] = to_bf16(acc);
    __syncthreads();
  }
  for (int s = kS; s < kSP; s++)
    qout[((size_t)b * kSP + s) * kH + t] = to_bf16(0.f);
}

// ---------------- K3: logits = k@q^T (WMMA) + softmax over S + colsum -----------------
// grid: (N/128, B) x 256 threads (8 waves, 16 rows each)
__global__ __launch_bounds__(256) void k_attn(
    const bf16_t* __restrict__ kmat, const bf16_t* __restrict__ qmat,
    bf16_t* __restrict__ attn_t, float* __restrict__ colsum,
    float* __restrict__ masks_out, int write_masks) {
  const int b = blockIdx.y;
  const int tid = threadIdx.x, wave = tid >> 5, lane = tid & 31;
  __shared__ float slog[8][16][16];
  __shared__ float csum[kSP];
  if (tid < kSP) csum[tid] = 0.f;

  const int n0  = blockIdx.x * 128 + wave * 16;
  const int m16 = lane & 15;
  const int kb  = (lane < 16) ? 0 : 8;
  v8f c = {0.f, 0.f, 0.f, 0.f, 0.f, 0.f, 0.f, 0.f};

  const bf16_t* arow = kmat + ((size_t)b * kN + n0 + m16) * kH;
  const bf16_t* brow = qmat + ((size_t)b * kSP + m16) * kH;
#pragma unroll
  for (int kk = 0; kk < kH; kk += 32) {
    v16bf a  = concat8(*(const v8bf*)&arow[kk + kb],
                       *(const v8bf*)&arow[kk + 16 + kb]);
    v16bf bq = concat8(*(const v8bf*)&brow[kk + kb],
                       *(const v8bf*)&brow[kk + 16 + kb]);
    c = __builtin_amdgcn_wmma_f32_16x16x32_bf16(false, a, false, bq, (short)0, c, false, false);
  }
  const int rbase = 8 * (lane >> 4);
#pragma unroll
  for (int i = 0; i < 8; i++) slog[wave][rbase + i][m16] = c[i];
  __syncthreads();

  if (tid < 128) {
    const int w2 = tid >> 4, rr = tid & 15;
    const int n = blockIdx.x * 128 + tid;
    float mx = -3.4e38f;
    for (int s = 0; s < kS; s++) mx = fmaxf(mx, slog[w2][rr][s]);
    float e[kS], sum = 0.f;
    for (int s = 0; s < kS; s++) { e[s] = __expf(slog[w2][rr][s] - mx); sum += e[s]; }
    float inv = 1.f / sum;
    for (int s = 0; s < kS; s++) {
      float p = e[s] * inv;                 // attn_vis
      if (write_masks) masks_out[((size_t)b * kS + s) * kN + n] = p;
      float ae = p + kEPS;
      attn_t[((size_t)b * kSP + s) * kN + n] = to_bf16(ae);
      atomicAdd(&csum[s], ae);
    }
  }
  __syncthreads();
  if (tid < kS) atomicAdd(&colsum[b * kSP + tid], csum[tid]);
}

// ---------------- K4: updates = attn^T @ v / colsum (WMMA over N) ---------------------
// grid: B blocks x 256 threads (8 waves -> 8 H-tiles of 16)
__global__ __launch_bounds__(256) void k_updates(
    const bf16_t* __restrict__ attn_t, const bf16_t* __restrict__ vt,
    const float* __restrict__ colsum, float* __restrict__ updates) {
  const int b = blockIdx.x;
  const int tid = threadIdx.x, wave = tid >> 5, lane = tid & 31;
  const int m16 = lane & 15;
  const int kb  = (lane < 16) ? 0 : 8;
  const int h0  = wave * 16;
  v8f c = {0.f, 0.f, 0.f, 0.f, 0.f, 0.f, 0.f, 0.f};

  const bf16_t* arow = attn_t + ((size_t)b * kSP + m16) * kN;
  const bf16_t* brow = vt + ((size_t)b * kH + h0 + m16) * kN;
  for (int n = 0; n < kN; n += 32) {
    if (n + 512 < kN) {                     // gfx1250 global_prefetch_b8 path
      __builtin_prefetch(&arow[n + 512], 0, 0);
      __builtin_prefetch(&brow[n + 512], 0, 0);
    }
    v16bf a  = concat8(*(const v8bf*)&arow[n + kb],
                       *(const v8bf*)&arow[n + 16 + kb]);
    v16bf bb = concat8(*(const v8bf*)&brow[n + kb],
                       *(const v8bf*)&brow[n + 16 + kb]);
    c = __builtin_amdgcn_wmma_f32_16x16x32_bf16(false, a, false, bb, (short)0, c, false, false);
  }
  const int rbase = 8 * (lane >> 4);
#pragma unroll
  for (int i = 0; i < 8; i++) {
    int s = rbase + i;
    float val = (s < kS) ? (c[i] / colsum[b * kSP + s]) : 0.f;
    updates[((size_t)b * kSP + s) * kH + h0 + m16] = val;
  }
}

// ---------------- K5: GRU cell + residual MLP (tiny, fp32 VALU) -----------------------
__global__ __launch_bounds__(256) void k_gru_mlp(
    const float* __restrict__ updates, float* __restrict__ slots_cur,
    const float* __restrict__ gru_wi, const float* __restrict__ gru_wh,
    const float* __restrict__ gru_bi, const float* __restrict__ gru_bh,
    const float* __restrict__ lnmg, const float* __restrict__ lnmb,
    const float* __restrict__ W1, const float* __restrict__ b1,
    const float* __restrict__ W2, const float* __restrict__ b2,
    float* __restrict__ slots_out, int is_last) {
  const int b = blockIdx.x, t = threadIdx.x;
  __shared__ float upd[kH], hprev[kH], gi[3 * kH], gh[3 * kH];
  __shared__ float snew[kH], lnh[kH], hid[kMLP];
  __shared__ float red1[128], red2[128];

  for (int s = 0; s < kS; s++) {
    if (t < kH) {
      upd[t]   = updates[((size_t)b * kSP + s) * kH + t];
      hprev[t] = slots_cur[((size_t)b * kS + s) * kH + t];
    }
    __syncthreads();
    for (int col = t; col < 3 * kH; col += 256) {
      float accI = gru_bi[col], accH = gru_bh[col];
      for (int j = 0; j < kH; j++) {
        accI += upd[j]   * gru_wi[(size_t)j * 3 * kH + col];
        accH += hprev[j] * gru_wh[(size_t)j * 3 * kH + col];
      }
      gi[col] = accI; gh[col] = accH;
    }
    __syncthreads();
    if (t < kH) {
      float r  = 1.f / (1.f + __expf(-(gi[t] + gh[t])));
      float z  = 1.f / (1.f + __expf(-(gi[kH + t] + gh[kH + t])));
      float nn = tanhf(gi[2 * kH + t] + r * gh[2 * kH + t]);
      float h  = (1.f - z) * nn + z * hprev[t];
      snew[t] = h;
      red1[t] = h; red2[t] = h * h;
    }
    __syncthreads();
    for (int off = 64; off > 0; off >>= 1) {
      if (t < off) { red1[t] += red1[t + off]; red2[t] += red2[t + off]; }
      __syncthreads();
    }
    float m = red1[0] / kH;
    float var = red2[0] / kH - m * m;
    float rstd = rsqrtf(var + 1e-5f);
    if (t < kH) lnh[t] = (snew[t] - m) * rstd * lnmg[t] + lnmb[t];
    __syncthreads();
    for (int col = t; col < kMLP; col += 256) {
      float acc = b1[col];
      for (int j = 0; j < kH; j++) acc += lnh[j] * W1[(size_t)j * kMLP + col];
      hid[col] = fmaxf(acc, 0.f);
    }
    __syncthreads();
    if (t < kH) {
      float acc = b2[t];
      for (int j = 0; j < kMLP; j++) acc += hid[j] * W2[(size_t)j * kH + t];
      float out = snew[t] + acc;
      slots_cur[((size_t)b * kS + s) * kH + t] = out;
      if (is_last) slots_out[((size_t)b * kS + s) * kH + t] = out;
    }
    __syncthreads();
  }
}

// ---------------- host-side orchestration ---------------------------------------------
extern "C" void kernel_launch(void* const* d_in, const int* in_sizes, int n_in,
                              void* d_out, int out_size, void* d_ws, size_t ws_size,
                              hipStream_t stream) {
  (void)in_sizes; (void)n_in; (void)out_size; (void)ws_size;

  const float* slots    = (const float*)d_in[0];
  const float* features = (const float*)d_in[1];
  const float* ln_in_g  = (const float*)d_in[2];
  const float* ln_in_b  = (const float*)d_in[3];
  const float* ln_sl_g  = (const float*)d_in[4];
  const float* ln_sl_b  = (const float*)d_in[5];
  const float* ln_ml_g  = (const float*)d_in[6];
  const float* ln_ml_b  = (const float*)d_in[7];
  const float* Wq       = (const float*)d_in[8];
  const float* Wk       = (const float*)d_in[9];
  const float* Wv       = (const float*)d_in[10];
  const float* gru_wi   = (const float*)d_in[11];
  const float* gru_wh   = (const float*)d_in[12];
  const float* gru_bi   = (const float*)d_in[13];
  const float* gru_bh   = (const float*)d_in[14];
  const float* W1       = (const float*)d_in[15];
  const float* b1       = (const float*)d_in[16];
  const float* W2       = (const float*)d_in[17];
  const float* b2       = (const float*)d_in[18];

  uint8_t* ws = (uint8_t*)d_ws;
  bf16_t* kbuf   = (bf16_t*)(ws + OFF_K);
  bf16_t* vtbuf  = (bf16_t*)(ws + OFF_VT);
  bf16_t* wkb    = (bf16_t*)(ws + OFF_WK);
  bf16_t* wvb    = (bf16_t*)(ws + OFF_WV);
  bf16_t* qbuf   = (bf16_t*)(ws + OFF_Q);
  bf16_t* attn_t = (bf16_t*)(ws + OFF_AT);
  float*  colsum = (float*)(ws + OFF_CS);
  float*  upd    = (float*)(ws + OFF_UPD);
  float*  slcur  = (float*)(ws + OFF_SL);

  float* slots_out = (float*)d_out;                        // [B,S,H]
  float* masks_out = (float*)d_out + (size_t)kB * kS * kH; // [B,S,N]

  hipMemcpyAsync(slcur, slots, (size_t)kB * kS * kH * sizeof(float),
                 hipMemcpyDeviceToDevice, stream);

  k_wconv<<<(kD * kH + 255) / 256, 256, 0, stream>>>(Wk, Wv, wkb, wvb);
  k_ln_project<<<kB * kN / 16, 256, 0, stream>>>(features, ln_in_g, ln_in_b,
                                                 wkb, wvb, kbuf, vtbuf);

  for (int it = 0; it < 3; ++it) {
    const int last = (it == 2);
    k_qproj<<<kB, 128, 0, stream>>>(slcur, ln_sl_g, ln_sl_b, Wq, qbuf, colsum);
    k_attn<<<dim3(kN / 128, kB), 256, 0, stream>>>(kbuf, qbuf, attn_t, colsum,
                                                   masks_out, last);
    k_updates<<<kB, 256, 0, stream>>>(attn_t, vtbuf, colsum, upd);
    k_gru_mlp<<<kB, 256, 0, stream>>>(upd, slcur, gru_wi, gru_wh, gru_bi, gru_bh,
                                      ln_ml_g, ln_ml_b, W1, b1, W2, b2,
                                      slots_out, last);
  }
}
